// SelfAttention_80547816669523
// MI455X (gfx1250) — matrix operand order repaired and checked by reading, not compiled, not run on hardware
//
#include <hip/hip_runtime.h>
#include <hip/hip_bf16.h>

#define B_   4
#define T_   4096
#define DIN  1024
#define H_   64
#define NEG_BIG (-3.0e38f)

typedef __attribute__((ext_vector_type(16))) _Float16 v16h;
typedef __attribute__((ext_vector_type(8)))  float    v8f;

// ---------------------------------------------------------------------------
// Kernel 0: one-shot conversion of Wq/Wk/Wv (fp32) -> f16 workspace copies.
// ---------------------------------------------------------------------------
__global__ __launch_bounds__(256) void cvt_w_kernel(
    const float* __restrict__ Wq, const float* __restrict__ Wk,
    const float* __restrict__ Wv, _Float16* __restrict__ wh)
{
    const int idx = blockIdx.x * 256 + threadIdx.x;      // [0, 3*64*1024)
    const int off = idx & (64 * DIN - 1);
    const float* src = (idx < 64 * DIN) ? Wq : (idx < 2 * 64 * DIN ? Wk : Wv);
    wh[idx] = (_Float16)src[off];
}

// ---------------------------------------------------------------------------
// Kernel 1: fused QKV projection. One wave = 16-row x 64-col tile of q,k,v.
// All 12 B-fragments are loaded into distinct registers per K-step (one load
// clause), then 12 WMMAs issue back-to-back -> loads pipeline with WMMA.
// q,k stored [B,T,H]; v stored transposed [B,H,T].
// ---------------------------------------------------------------------------
__global__ __launch_bounds__(128) void qkv_proj_kernel(
    const float* __restrict__ x,
    const _Float16* __restrict__ wh,               // [3][64][1024] f16
    const float* __restrict__ bq, const float* __restrict__ bk,
    const float* __restrict__ bv,
    _Float16* __restrict__ qf, _Float16* __restrict__ kf, _Float16* __restrict__ vT)
{
    const int lane = threadIdx.x & 31;
    const int wave = threadIdx.x >> 5;
    const int tile = blockIdx.x * 4 + wave;          // [0, B*T/16)
    const int b    = tile / (T_ / 16);
    const int i0   = (tile % (T_ / 16)) * 16;

    const int arow    = lane & 15;                   // A-frag: M row
    const int khalf_a = (lane < 16) ? 0 : 16;        // A-frag K sub-pattern
    const int kbase_b = (lane < 16) ? 0 : 16;        // B-frag K sub-pattern
    const int ncol    = lane & 15;                   // B/C/D column
    const int rbase   = (lane >> 4) * 8;             // C/D row base

    v8f acc[3][4];
    #pragma unroll
    for (int m = 0; m < 3; ++m)
        #pragma unroll
        for (int nt = 0; nt < 4; ++nt)
            #pragma unroll
            for (int r = 0; r < 8; ++r) acc[m][nt][r] = 0.0f;

    const float*    xbase = x + ((size_t)b * T_ + i0 + arow) * DIN;
    const _Float16* wbase = wh + (size_t)ncol * DIN + kbase_b;

    for (int k0 = 0; k0 < DIN; k0 += 32) {
        // --- load ALL 12 B fragments first (distinct regs, one clause) ---
        v16h bf[12];
        #pragma unroll
        for (int m = 0; m < 3; ++m)
            #pragma unroll
            for (int nt = 0; nt < 4; ++nt)
                bf[m * 4 + nt] = *(const v16h*)(wbase + ((size_t)m * 64 + nt * 16) * DIN + k0);

        // --- A fragment: x rows, fp32 -> f16 inline ---
        v16h afrag;
        const float* pa = xbase + k0 + khalf_a;
        #pragma unroll
        for (int i = 0; i < 8; ++i) {
            afrag[i]     = (_Float16)pa[i];
            afrag[8 + i] = (_Float16)pa[8 + i];
        }

        // --- 12 WMMAs back-to-back ---
        #pragma unroll
        for (int m = 0; m < 3; ++m)
            #pragma unroll
            for (int nt = 0; nt < 4; ++nt)
                acc[m][nt] = __builtin_amdgcn_wmma_f32_16x16x32_f16(
                    false, afrag, false, bf[m * 4 + nt], (short)0, acc[m][nt], false, false);
    }

    // --- epilogue: bias add, f16 stores ---
    const float* biasm[3] = {bq, bk, bv};
    #pragma unroll
    for (int m = 0; m < 3; ++m) {
        #pragma unroll
        for (int nt = 0; nt < 4; ++nt) {
            const int h    = nt * 16 + ncol;
            const float bb = biasm[m][h];
            #pragma unroll
            for (int r = 0; r < 8; ++r) {
                const int row = i0 + rbase + r;
                const _Float16 val = (_Float16)(acc[m][nt][r] + bb);
                if (m == 0)      qf[((size_t)b * T_ + row) * H_ + h] = val;
                else if (m == 1) kf[((size_t)b * T_ + row) * H_ + h] = val;
                else             vT[((size_t)b * H_ + h) * T_ + row] = val;  // transposed
            }
        }
    }
}

// ---------------------------------------------------------------------------
// Kernel 2: flash attention. Rows = K positions (reference: aff = K.Q^T,
// tril mask, softmax over cols, then .V). One wave owns 16 rows; each
// iteration: load q+v fragments up front (latency hides under WMMA/softmax),
// 4 WMMA for S, online softmax, LDS transpose of P, 4 WMMA for O.
// ---------------------------------------------------------------------------
__global__ __launch_bounds__(128) void attn_kernel(
    const _Float16* __restrict__ qf, const _Float16* __restrict__ kf,
    const _Float16* __restrict__ vT, float* __restrict__ out)
{
    __shared__ _Float16 plds[4][16][48];   // per-wave P staging (16x32, padded)

    const int lane = threadIdx.x & 31;
    const int wave = threadIdx.x >> 5;
    const int tile = blockIdx.x * 4 + wave;
    const int b    = tile / (T_ / 16);
    const int i0   = (tile % (T_ / 16)) * 16;

    const int arow    = lane & 15;
    const int khalf_a = (lane < 16) ? 0 : 16;
    const int kbase_b = (lane < 16) ? 0 : 16;
    const int ncol    = lane & 15;
    const int rbase   = (lane >> 4) * 8;

    const _Float16* kb = kf + (size_t)b * T_ * H_;
    const _Float16* qb = qf + (size_t)b * T_ * H_;
    const _Float16* vb = vT + (size_t)b * H_ * T_;

    // A fragments for the 16 attention rows (k rows i0..i0+15), H=64 = 2 chunks
    v16h aq[2];
    #pragma unroll
    for (int c = 0; c < 2; ++c) {
        const _Float16* pa = kb + (size_t)(i0 + arow) * H_ + c * 32 + khalf_a;
        #pragma unroll
        for (int i = 0; i < 8; ++i) {
            aq[c][i]     = pa[i];
            aq[c][8 + i] = pa[8 + i];
        }
    }

    v8f   o[4];
    float mrow[8], lrow[8];
    #pragma unroll
    for (int r = 0; r < 8; ++r) { mrow[r] = NEG_BIG; lrow[r] = 0.0f; }
    #pragma unroll
    for (int nt = 0; nt < 4; ++nt)
        #pragma unroll
        for (int r = 0; r < 8; ++r) o[nt][r] = 0.0f;

    const int jend = i0 + 16;               // causal: j <= i, max i = i0+15
    for (int j0 = 0; j0 < jend; j0 += 32) {
        // ---- load ALL fragments for this block up front (q for S, v for O) ----
        v16h qfr[2][2], vfr[4];
        #pragma unroll
        for (int t = 0; t < 2; ++t)
            #pragma unroll
            for (int c = 0; c < 2; ++c)
                qfr[t][c] = *(const v16h*)(qb + (size_t)(j0 + t * 16 + ncol) * H_
                                              + c * 32 + kbase_b);
        #pragma unroll
        for (int nt = 0; nt < 4; ++nt)
            vfr[nt] = *(const v16h*)(vb + (size_t)(nt * 16 + ncol) * T_ + j0 + kbase_b);

        // prefetch next j-block's q and vT lines (L2-resident)
        if (j0 + 32 < jend) {
            __builtin_prefetch(qb + (size_t)(j0 + 32 + ncol) * H_, 0, 0);
            __builtin_prefetch(vb + (size_t)ncol * T_ + j0 + 32, 0, 0);
        }

        // ---- S : two 16-wide j tiles, K=64 over 2 chunks of 32 ----
        v8f s[2];
        #pragma unroll
        for (int t = 0; t < 2; ++t) {
            #pragma unroll
            for (int r = 0; r < 8; ++r) s[t][r] = 0.0f;
            #pragma unroll
            for (int c = 0; c < 2; ++c)
                s[t] = __builtin_amdgcn_wmma_f32_16x16x32_f16(
                    false, aq[c], false, qfr[t][c], (short)0, s[t], false, false);
        }

        // ---- causal mask (diagonal blocks only) ----
        if (j0 + 31 > i0) {
            #pragma unroll
            for (int t = 0; t < 2; ++t)
                #pragma unroll
                for (int r = 0; r < 8; ++r) {
                    const int i = i0 + rbase + r;
                    const int j = j0 + t * 16 + ncol;
                    if (j > i) s[t][r] = NEG_BIG;
                }
        }

        // ---- online softmax: row reductions across 16-lane half-wave ----
        float rmax[8];
        #pragma unroll
        for (int r = 0; r < 8; ++r) rmax[r] = fmaxf(s[0][r], s[1][r]);
        #pragma unroll
        for (int off = 1; off < 16; off <<= 1)
            #pragma unroll
            for (int r = 0; r < 8; ++r)
                rmax[r] = fmaxf(rmax[r], __shfl_xor(rmax[r], off));

        float scale[8];
        #pragma unroll
        for (int r = 0; r < 8; ++r) {
            const float mnew = fmaxf(mrow[r], rmax[r]);
            scale[r] = __expf(mrow[r] - mnew);
            mrow[r]  = mnew;
            lrow[r] *= scale[r];
        }
        #pragma unroll
        for (int nt = 0; nt < 4; ++nt)
            #pragma unroll
            for (int r = 0; r < 8; ++r) o[nt][r] *= scale[r];

        // ---- P = exp(S - m), row sums, stage to LDS for layout transpose ----
        float rsum[8];
        #pragma unroll
        for (int r = 0; r < 8; ++r) rsum[r] = 0.0f;
        #pragma unroll
        for (int t = 0; t < 2; ++t)
            #pragma unroll
            for (int r = 0; r < 8; ++r) {
                const float p = __expf(s[t][r] - mrow[r]);
                rsum[r] += p;
                plds[wave][rbase + r][t * 16 + ncol] = (_Float16)p;
            }
        #pragma unroll
        for (int off = 1; off < 16; off <<= 1)
            #pragma unroll
            for (int r = 0; r < 8; ++r)
                rsum[r] += __shfl_xor(rsum[r], off);
        #pragma unroll
        for (int r = 0; r < 8; ++r) lrow[r] += rsum[r];

        // ---- reload P as an A fragment (16x32, K = 32 keys) ----
        v16h pfrag;
        {
            const _Float16* pp = &plds[wave][arow][khalf_a];
            #pragma unroll
            for (int i = 0; i < 8; ++i) {
                pfrag[i]     = pp[i];
                pfrag[8 + i] = pp[8 + i];
            }
        }

        // ---- O += P x V (v fragments already in registers) ----
        #pragma unroll
        for (int nt = 0; nt < 4; ++nt)
            o[nt] = __builtin_amdgcn_wmma_f32_16x16x32_f16(
                false, pfrag, false, vfr[nt], (short)0, o[nt], false, false);
    }

    // ---- normalize and store fp32 output [B,T,H] ----
    float rinv[8];
    #pragma unroll
    for (int r = 0; r < 8; ++r) rinv[r] = 1.0f / lrow[r];
    float* ob = out + ((size_t)b * T_ + i0) * H_;
    #pragma unroll
    for (int nt = 0; nt < 4; ++nt)
        #pragma unroll
        for (int r = 0; r < 8; ++r)
            ob[(size_t)(rbase + r) * H_ + nt * 16 + ncol] = o[nt][r] * rinv[r];
}

// ---------------------------------------------------------------------------
extern "C" void kernel_launch(void* const* d_in, const int* in_sizes, int n_in,
                              void* d_out, int out_size, void* d_ws, size_t ws_size,
                              hipStream_t stream) {
    const float* x  = (const float*)d_in[0];
    const float* Wq = (const float*)d_in[1];
    const float* bq = (const float*)d_in[2];
    const float* Wk = (const float*)d_in[3];
    const float* bk = (const float*)d_in[4];
    const float* Wv = (const float*)d_in[5];
    const float* bv = (const float*)d_in[6];
    float* out = (float*)d_out;

    const size_t nqkv = (size_t)B_ * T_ * H_;          // 1,048,576 halves each
    _Float16* qf = (_Float16*)d_ws;
    _Float16* kf = qf + nqkv;
    _Float16* vT = kf + nqkv;                          // [B][H][T]
    _Float16* wh = vT + nqkv;                          // [3][64][1024]

    cvt_w_kernel<<<(3 * 64 * DIN) / 256, 256, 0, stream>>>(Wq, Wk, Wv, wh);

    const int tiles = B_ * (T_ / 16);                  // 1024 row-tiles
    qkv_proj_kernel<<<tiles / 4, 128, 0, stream>>>(x, wh, bq, bk, bv, qf, kf, vT);
    attn_kernel<<<tiles / 4, 128, 0, stream>>>(qf, kf, vT, out);
}
